// gruCrf_9182640079406
// MI455X (gfx1250) — matrix operand order repaired and checked by reading, not compiled, not run on hardware
//
#include <hip/hip_runtime.h>
#include <hip/hip_bf16.h>

// ---------------- problem constants ----------------
constexpr int V_  = 21128;
constexpr int H_  = 512;
constexpr int T_  = 9;
constexpr int B_  = 64;
constexpr int S_  = 512;
constexpr int Hd_ = 256;   // H/2
constexpr int G_  = 768;   // 3*Hd

typedef __bf16 bf16;
typedef __attribute__((ext_vector_type(16))) __bf16 v16bf;
typedef __attribute__((ext_vector_type(8)))  __bf16 v8bf;
typedef __attribute__((ext_vector_type(8)))  float  v8f;

__device__ __forceinline__ v8f wmma_bf16(v16bf a, v16bf b, v8f c) {
  return __builtin_amdgcn_wmma_f32_16x16x32_bf16(
      /*neg_a=*/false, a, /*neg_b=*/false, b,
      /*c_mod=*/(short)0, c, /*reuse_a=*/false, /*reuse_b=*/false);
}

// Build A-tile per-lane vector from two contiguous 8-element chunks.
// 16-bit A 16x32 layout: lane(half,m); elems 0..7 -> K = 8*half + e,
// elems 8..15 -> K = 16 + 8*half + (e-8).
__device__ __forceinline__ v16bf mk_a(v8bf lo, v8bf hi) {
  v16bf a;
#pragma unroll
  for (int e = 0; e < 8; ++e) { a[e] = lo[e]; a[e + 8] = hi[e]; }
  return a;
}

// ---------------- workspace layout (bytes, all 256-aligned) ----------------
constexpr size_t SZ_X    = (size_t)S_ * B_ * H_ * 2;     // x bf16 [S*B][512]
constexpr size_t SZ_WIH  = (size_t)G_ * H_ * 2;          // 768x512 bf16
constexpr size_t SZ_WHH  = (size_t)G_ * Hd_ * 2;         // 768x256 bf16
constexpr size_t SZ_OUTW = (size_t)16 * H_ * 2;          // padded 16x512 bf16
constexpr size_t SZ_XW   = (size_t)S_ * B_ * G_ * 4;     // f32 [S*B][768]
constexpr size_t SZ_HS   = (size_t)S_ * B_ * Hd_ * 4;    // f32 [S*B][256]
constexpr size_t SZ_LOG  = (size_t)S_ * B_ * 16 * 4;     // f32 [S*B][16]

constexpr size_t OFF_X    = 0;
constexpr size_t OFF_WIHF = OFF_X    + SZ_X;
constexpr size_t OFF_WIHB = OFF_WIHF + SZ_WIH;
constexpr size_t OFF_WHHF = OFF_WIHB + SZ_WIH;
constexpr size_t OFF_WHHB = OFF_WHHF + SZ_WHH;
constexpr size_t OFF_OUTW = OFF_WHHB + SZ_WHH;
constexpr size_t OFF_XWF  = OFF_OUTW + SZ_OUTW;
constexpr size_t OFF_XWB  = OFF_XWF  + SZ_XW;
constexpr size_t OFF_HF   = OFF_XWB  + SZ_XW;
constexpr size_t OFF_HB   = OFF_HF   + SZ_HS;
constexpr size_t OFF_LOG  = OFF_HB   + SZ_HS;
constexpr size_t OFF_HIST = OFF_LOG  + SZ_LOG;           // int32 [S-1][B][T]

// ---------------- weight prep: f32 -> bf16 (and pad outW to 16 rows) -------
__global__ void prep_weights(const float* __restrict__ WihF,
                             const float* __restrict__ WihB,
                             const float* __restrict__ WhhF,
                             const float* __restrict__ WhhB,
                             const float* __restrict__ outW,
                             bf16* wihF, bf16* wihB, bf16* whhF, bf16* whhB,
                             bf16* outWp) {
  const int nWih = G_ * H_;     // 393216
  const int nWhh = G_ * Hd_;    // 196608
  const int nOut = 16 * H_;     // 8192
  const int total = 2 * nWih + 2 * nWhh + nOut;
  for (int i = blockIdx.x * blockDim.x + threadIdx.x; i < total;
       i += gridDim.x * blockDim.x) {
    int idx = i;
    if (idx < nWih)               { wihF[idx] = (bf16)WihF[idx]; continue; }
    idx -= nWih;
    if (idx < nWih)               { wihB[idx] = (bf16)WihB[idx]; continue; }
    idx -= nWih;
    if (idx < nWhh)               { whhF[idx] = (bf16)WhhF[idx]; continue; }
    idx -= nWhh;
    if (idx < nWhh)               { whhB[idx] = (bf16)WhhB[idx]; continue; }
    idx -= nWhh;
    int row = idx / H_, col = idx % H_;
    outWp[idx] = (row < T_) ? (bf16)outW[row * H_ + col] : (bf16)0.0f;
  }
}

// ---------------- embedding gather: x[s*B+b][k] = emb[src[b][s]][k] --------
__global__ void embed_gather(const int* __restrict__ src,
                             const float* __restrict__ emb,
                             bf16* __restrict__ X) {
  // one thread per 4 consecutive k
  int i = blockIdx.x * blockDim.x + threadIdx.x;      // 0 .. S*B*128-1
  if (i >= S_ * B_ * (H_ / 4)) return;
  int row = i >> 7;                  // s*B+b
  int k4  = (i & 127) << 2;
  int s = row / B_, b = row % B_;
  int tok = src[b * S_ + s];
  const float4 v = *(const float4*)(emb + (size_t)tok * H_ + k4);
  bf16* dst = X + (size_t)row * H_ + k4;
  dst[0] = (bf16)v.x; dst[1] = (bf16)v.y; dst[2] = (bf16)v.z; dst[3] = (bf16)v.w;
}

// ---------------- projection GEMM: Out[M,N] = X[M,K] * W[N,K]^T + bias -----
// 64x64 macro-tile per wave: 4 A-tiles x 4 B-tiles -> 16 WMMAs per K-step.
// Arithmetic intensity ~4x over a single-tile wave; 8 waves / block.
__global__ __launch_bounds__(256) void proj_gemm(
    const bf16* __restrict__ X,
    const bf16* __restrict__ W,
    const float* __restrict__ bias,
    float* __restrict__ Out,
    int M, int N, int K) {
  int wid  = (blockIdx.x * blockDim.x + threadIdx.x) >> 5;
  int lane = threadIdx.x & 31;
  int mblk = M >> 6, nblk = N >> 6;
  int mb = wid % mblk, nb = wid / mblk;
  if (nb >= nblk) return;            // wave-uniform
  int hl = lane >> 4, l16 = lane & 15;

  const bf16* arow[4];
  const bf16* brow[4];
#pragma unroll
  for (int i = 0; i < 4; ++i) {
    arow[i] = X + (size_t)(mb * 64 + i * 16 + l16) * K;
    brow[i] = W + (size_t)(nb * 64 + i * 16 + l16) * K;
  }

  v8f acc[4][4];
#pragma unroll
  for (int mi = 0; mi < 4; ++mi)
#pragma unroll
    for (int ni = 0; ni < 4; ++ni) acc[mi][ni] = (v8f){};

  for (int kb = 0; kb < K; kb += 32) {
    v16bf a[4], b[4];
#pragma unroll
    for (int i = 0; i < 4; ++i) {
      v8bf a0 = *(const v8bf*)(arow[i] + kb + 8 * hl);
      v8bf a1 = *(const v8bf*)(arow[i] + kb + 16 + 8 * hl);
      a[i] = mk_a(a0, a1);
      b[i] = mk_a(*(const v8bf*)(brow[i] + kb + 8 * hl), *(const v8bf*)(brow[i] + kb + 16 + 8 * hl));
    }
#pragma unroll
    for (int mi = 0; mi < 4; ++mi)
#pragma unroll
      for (int ni = 0; ni < 4; ++ni)
        acc[mi][ni] = wmma_bf16(a[mi], b[ni], acc[mi][ni]);
  }

#pragma unroll
  for (int ni = 0; ni < 4; ++ni) {
    int n = nb * 64 + ni * 16 + l16;
    float bcol = bias[n];
#pragma unroll
    for (int mi = 0; mi < 4; ++mi) {
#pragma unroll
      for (int v = 0; v < 8; ++v) {
        int m = mb * 64 + mi * 16 + v + 8 * hl;
        Out[(size_t)m * N + n] = acc[mi][ni][v] + bcol;
      }
    }
  }
}

// ---------------- persistent GRU recurrence (one block per direction) ------
// h kept in LDS as bf16 [64][256]; per step: hw = h*Whh^T via WMMA (A-tiles
// come from LDS via ds_load_b128, reused across 3 gate B-tiles), gates in
// VALU, h updated, hidden state streamed to global.
__global__ __launch_bounds__(512) void gru_rec(
    const float* __restrict__ xw_f, const float* __restrict__ xw_b,
    const bf16* __restrict__ WhhF,  const bf16* __restrict__ WhhB,
    const float* __restrict__ bhhF, const float* __restrict__ bhhB,
    float* __restrict__ hF, float* __restrict__ hB) {
  __shared__ __align__(32) bf16 hsh[B_ * Hd_];   // 32 KB

  const int dir = blockIdx.x;                    // 0 = fwd, 1 = bwd
  const float* xw  = dir ? xw_b : xw_f;
  const bf16*  Whh = dir ? WhhB : WhhF;
  const float* bhh = dir ? bhhB : bhhF;
  float*       hout = dir ? hB : hF;

  for (int i = threadIdx.x; i < B_ * Hd_; i += blockDim.x) hsh[i] = (bf16)0.0f;
  __syncthreads();

  const int wv   = threadIdx.x >> 5;   // wave id = column tile 0..15
  const int lane = threadIdx.x & 31;
  const int hl   = lane >> 4, l16 = lane & 15;
  const int ct   = wv;
  const int j    = ct * 16 + l16;      // hidden column handled by this lane
  const float br = bhh[j], bz = bhh[Hd_ + j], bn = bhh[2 * Hd_ + j];
  const bf16* bRr = Whh + (size_t)(0 * Hd_ + ct * 16 + l16) * Hd_;
  const bf16* bZr = Whh + (size_t)(1 * Hd_ + ct * 16 + l16) * Hd_;
  const bf16* bNr = Whh + (size_t)(2 * Hd_ + ct * 16 + l16) * Hd_;

  for (int step = 0; step < S_; ++step) {
    const int s = dir ? (S_ - 1 - step) : step;
    float hnew[4][8];

#pragma unroll
    for (int mt = 0; mt < 4; ++mt) {
      v8f aR = {}, aZ = {}, aN = {};
      const bf16* hrow = hsh + (size_t)(mt * 16 + l16) * Hd_;
      for (int kb = 0; kb < Hd_; kb += 32) {
        v8bf a0 = *(const v8bf*)(hrow + kb + 16 * hl);
        v8bf a1 = *(const v8bf*)(hrow + kb + 16 * hl + 8);
        v16bf a = mk_a(a0, a1);
        v16bf br_ = *(const v16bf*)(bRr + kb + 16 * hl);
        v16bf bz_ = *(const v16bf*)(bZr + kb + 16 * hl);
        v16bf bn_ = *(const v16bf*)(bNr + kb + 16 * hl);
        aR = wmma_bf16(a, br_, aR);
        aZ = wmma_bf16(a, bz_, aZ);
        aN = wmma_bf16(a, bn_, aN);
      }
#pragma unroll
      for (int v = 0; v < 8; ++v) {
        int b = mt * 16 + v + 8 * hl;
        const float* xr = xw + ((size_t)s * B_ + b) * G_;
        float xrv = xr[j], xzv = xr[Hd_ + j], xnv = xr[2 * Hd_ + j];
        float hprev = (float)hsh[b * Hd_ + j];
        float r = 1.0f / (1.0f + __expf(-(xrv + aR[v] + br)));
        float z = 1.0f / (1.0f + __expf(-(xzv + aZ[v] + bz)));
        float n = tanhf(xnv + r * (aN[v] + bn));
        hnew[mt][v] = (1.0f - z) * n + z * hprev;
      }
    }
    __syncthreads();   // all reads of h done
#pragma unroll
    for (int mt = 0; mt < 4; ++mt) {
#pragma unroll
      for (int v = 0; v < 8; ++v) {
        int b = mt * 16 + v + 8 * hl;
        hsh[b * Hd_ + j] = (bf16)hnew[mt][v];
        hout[((size_t)s * B_ + b) * Hd_ + j] = hnew[mt][v];
      }
    }
    __syncthreads();   // h updated for next step
  }
}

// ---------------- logits: [S*B,16] = concat(hf,hb)[S*B,512] * outWp^T ------
__global__ void logits_gemm(const float* __restrict__ hF,
                            const float* __restrict__ hB,
                            const bf16* __restrict__ outWp,
                            const float* __restrict__ outb,
                            float* __restrict__ logits) {
  int wid  = (blockIdx.x * blockDim.x + threadIdx.x) >> 5;
  int lane = threadIdx.x & 31;
  if (wid >= (S_ * B_) / 16) return;
  int hl = lane >> 4, l16 = lane & 15;
  const bf16* brow = outWp + (size_t)l16 * H_;
  v8f acc = {};
  for (int kb = 0; kb < H_; kb += 32) {
    const float* src = (kb < Hd_)
        ? (hF + (size_t)(wid * 16 + l16) * Hd_ + kb)
        : (hB + (size_t)(wid * 16 + l16) * Hd_ + (kb - Hd_));
    v16bf a;
#pragma unroll
    for (int e = 0; e < 8; ++e) {
      a[e]     = (bf16)src[8 * hl + e];
      a[e + 8] = (bf16)src[16 + 8 * hl + e];
    }
    v16bf b = mk_a(*(const v8bf*)(brow + kb + 8 * hl), *(const v8bf*)(brow + kb + 16 + 8 * hl));
    acc = wmma_bf16(a, b, acc);
  }
  float bias = (l16 < T_) ? outb[l16] : 0.0f;
#pragma unroll
  for (int v = 0; v < 8; ++v) {
    int m = wid * 16 + v + 8 * hl;
    logits[(size_t)m * 16 + l16] = acc[v] + bias;
  }
}

// ---------------- CRF log-likelihood -> loss ------------------------------
__global__ __launch_bounds__(576) void crf_llh(
    const float* __restrict__ logits, const float* __restrict__ start,
    const float* __restrict__ endt,   const float* __restrict__ trans,
    const int* __restrict__ label,    float* __restrict__ out) {
  __shared__ float sc[B_][T_];
  __shared__ float part[B_];
  const int t = threadIdx.x;
  const int b = t / T_, tp = t % T_;
  const bool act = t < B_ * T_;

  if (act) sc[b][tp] = start[tp] + logits[(size_t)b * 16 + tp];
  __syncthreads();
  for (int s = 1; s < S_; ++s) {
    float nv = 0.0f;
    if (act) {
      float m = -1e30f;
      for (int tt = 0; tt < T_; ++tt)
        m = fmaxf(m, sc[b][tt] + trans[tt * T_ + tp]);
      float sum = 0.0f;
      for (int tt = 0; tt < T_; ++tt)
        sum += __expf(sc[b][tt] + trans[tt * T_ + tp] - m);
      nv = m + __logf(sum) + logits[((size_t)s * B_ + b) * 16 + tp];
    }
    __syncthreads();
    if (act) sc[b][tp] = nv;
    __syncthreads();
  }
  if (t < B_) {
    const int bb = t;
    float m = -1e30f;
    for (int tt = 0; tt < T_; ++tt) m = fmaxf(m, sc[bb][tt] + endt[tt]);
    float sum = 0.0f;
    for (int tt = 0; tt < T_; ++tt) sum += __expf(sc[bb][tt] + endt[tt] - m);
    float den = m + __logf(sum);
    int prev = label[bb * S_];
    float num = start[prev];
    for (int s = 0; s < S_; ++s) {
      int tg = label[bb * S_ + s];
      num += logits[((size_t)s * B_ + bb) * 16 + tg];
      if (s > 0) num += trans[prev * T_ + tg];
      prev = tg;
    }
    num += endt[prev];
    part[bb] = num - den;
  }
  __syncthreads();
  if (t == 0) {
    float llh = 0.0f;
    for (int i = 0; i < B_; ++i) llh += part[i];
    out[0] = -llh / (float)B_;
  }
}

// ---------------- Viterbi decode + correct count + flat outputs ------------
__global__ __launch_bounds__(576) void viterbi(
    const float* __restrict__ logits, const float* __restrict__ start,
    const float* __restrict__ endt,   const float* __restrict__ trans,
    const int* __restrict__ label,    int* __restrict__ hist,
    float* __restrict__ out) {
  __shared__ float sc[B_][T_];
  __shared__ int cnt[B_];
  const int t = threadIdx.x;
  const int b = t / T_, tp = t % T_;
  const bool act = t < B_ * T_;

  if (act) sc[b][tp] = start[tp] + logits[(size_t)b * 16 + tp];
  __syncthreads();
  for (int s = 1; s < S_; ++s) {
    float nv = 0.0f;
    if (act) {
      float best = -1e30f; int arg = 0;
      for (int tt = 0; tt < T_; ++tt) {
        float c = sc[b][tt] + trans[tt * T_ + tp];
        if (c > best) { best = c; arg = tt; }
      }
      nv = best + logits[((size_t)s * B_ + b) * 16 + tp];
      hist[((size_t)(s - 1) * B_ + b) * T_ + tp] = arg;
    }
    __syncthreads();
    if (act) sc[b][tp] = nv;
    __syncthreads();
  }
  if (t < B_) {
    const int bb = t;
    int tag = 0; float bv = -1e30f;
    for (int tt = 0; tt < T_; ++tt) {
      float c = sc[bb][tt] + endt[tt];
      if (c > bv) { bv = c; tag = tt; }
    }
    int correct = 0;
    for (int s = S_ - 1; s >= 0; --s) {
      int lab = label[bb * S_ + s];
      int pred = (lab > 0) ? tag : 0;
      out[2 + bb * S_ + s]            = (float)pred;
      out[2 + S_ * B_ + bb * S_ + s]  = (float)lab;
      correct += (pred == lab) ? 1 : 0;
      if (s > 0) tag = hist[((size_t)(s - 1) * B_ + bb) * T_ + tag];
    }
    cnt[bb] = correct;
  }
  __syncthreads();
  if (t == 0) {
    int c = 0;
    for (int i = 0; i < B_; ++i) c += cnt[i];
    out[1] = (float)c;
  }
}

// ---------------- host orchestration --------------------------------------
extern "C" void kernel_launch(void* const* d_in, const int* in_sizes, int n_in,
                              void* d_out, int out_size, void* d_ws, size_t ws_size,
                              hipStream_t stream) {
  const int*   src    = (const int*)  d_in[0];
  const int*   label  = (const int*)  d_in[1];
  const float* emb    = (const float*)d_in[2];
  const float* WihF   = (const float*)d_in[3];
  const float* WhhF   = (const float*)d_in[4];
  const float* bihF   = (const float*)d_in[5];
  const float* bhhF   = (const float*)d_in[6];
  const float* WihB   = (const float*)d_in[7];
  const float* WhhB   = (const float*)d_in[8];
  const float* bihB   = (const float*)d_in[9];
  const float* bhhB   = (const float*)d_in[10];
  const float* outW   = (const float*)d_in[11];
  const float* outb   = (const float*)d_in[12];
  const float* startT = (const float*)d_in[13];
  const float* endT   = (const float*)d_in[14];
  const float* trans  = (const float*)d_in[15];
  float* out = (float*)d_out;

  char* ws = (char*)d_ws;
  bf16*  X     = (bf16*) (ws + OFF_X);
  bf16*  wihF  = (bf16*) (ws + OFF_WIHF);
  bf16*  wihB  = (bf16*) (ws + OFF_WIHB);
  bf16*  whhF  = (bf16*) (ws + OFF_WHHF);
  bf16*  whhB  = (bf16*) (ws + OFF_WHHB);
  bf16*  outWp = (bf16*) (ws + OFF_OUTW);
  float* xwF   = (float*)(ws + OFF_XWF);
  float* xwB   = (float*)(ws + OFF_XWB);
  float* hF    = (float*)(ws + OFF_HF);
  float* hB    = (float*)(ws + OFF_HB);
  float* logit = (float*)(ws + OFF_LOG);
  int*   hist  = (int*)  (ws + OFF_HIST);

  // 1) weights -> bf16 (+ pad outW)
  prep_weights<<<2048, 256, 0, stream>>>(WihF, WihB, WhhF, WhhB, outW,
                                         wihF, wihB, whhF, whhB, outWp);
  // 2) embedding gather -> x bf16 [S*B][512]
  embed_gather<<<(S_ * B_ * (H_ / 4) + 255) / 256, 256, 0, stream>>>(src, emb, X);
  // 3) input projections (bias fused): xw = x*Wih^T + bih, [S*B][768]
  {
    int waves  = (S_ * B_ / 64) * (G_ / 64);     // 512 * 12 = 6144
    int blocks = waves * 32 / 256;               // 768
    proj_gemm<<<blocks, 256, 0, stream>>>(X, wihF, bihF, xwF, S_ * B_, G_, H_);
    proj_gemm<<<blocks, 256, 0, stream>>>(X, wihB, bihB, xwB, S_ * B_, G_, H_);
  }
  // 4) sequential GRU, one persistent block per direction
  gru_rec<<<2, 512, 0, stream>>>(xwF, xwB, whhF, whhB, bhhF, bhhB, hF, hB);
  // 5) logits [S*B][16] (cols 0..8 valid)
  logits_gemm<<<(S_ * B_ / 16) * 32 / 256, 256, 0, stream>>>(hF, hB, outWp, outb, logit);
  // 6) CRF forward -> loss (out[0])
  crf_llh<<<1, 576, 0, stream>>>(logit, startT, endT, trans, label, out);
  // 7) Viterbi -> correct (out[1]) + predict_flat + label_flat
  viterbi<<<1, 576, 0, stream>>>(logit, startT, endT, trans, label, hist, out);

  (void)in_sizes; (void)n_in; (void)out_size; (void)ws_size;
}
